// FeedForwardNetwork_37022618091817
// MI455X (gfx1250) — 8-bit matrix operand order repaired and checked by reading, compiled, not run on hardware
//
#include <hip/hip_runtime.h>
#include <hip/hip_bf16.h>
#include <stdint.h>

typedef int v8i __attribute__((ext_vector_type(8)));
typedef int vi4 __attribute__((__vector_size__(16)));

#define N_TOK 4096
#define D_DIM 4096
#define F_DIM 11008
#define TSTRIDE 80

#if __has_builtin(__builtin_amdgcn_global_load_async_to_lds_b128)
#define HAVE_ASYNC_LDS 1
#else
#define HAVE_ASYNC_LDS 0
#endif

__device__ __forceinline__ void wait_async0() {
#if HAVE_ASYNC_LDS
#if __has_builtin(__builtin_amdgcn_s_wait_asynccnt)
  __builtin_amdgcn_s_wait_asynccnt(0);
#else
  asm volatile("s_wait_asynccnt 0x0" ::: "memory");
#endif
#endif
}

#if HAVE_ASYNC_LDS
__device__ __forceinline__ void async_cp16(void* ldsdst, const void* gsrc) {
  __builtin_amdgcn_global_load_async_to_lds_b128(
      (__attribute__((address_space(1))) vi4*)const_cast<void*>(gsrc),
      (__attribute__((address_space(3))) vi4*)ldsdst, 0, 0);
}
#endif

__device__ __forceinline__ v8i vzero8() {
  v8i z;
#pragma unroll
  for (int j = 0; j < 8; ++j) z[j] = 0;
  return z;
}

__global__ void k_init_slots(unsigned* slots) {
  if (threadIdx.x < 8) slots[threadIdx.x] = 0u;
}

__global__ __launch_bounds__(256) void k_absmax(const float4* __restrict__ src,
                                                long n4, unsigned* __restrict__ slot) {
  long i = (long)blockIdx.x * blockDim.x + threadIdx.x;
  long stride = (long)gridDim.x * blockDim.x;
  float m = 0.f;
  for (; i < n4; i += stride) {
    float4 v = src[i];
    m = fmaxf(m, fmaxf(fmaxf(fabsf(v.x), fabsf(v.y)), fmaxf(fabsf(v.z), fabsf(v.w))));
  }
#pragma unroll
  for (int off = 16; off > 0; off >>= 1)
    m = fmaxf(m, __shfl_xor(m, off, 32));
  if ((threadIdx.x & 31) == 0)
    atomicMax(slot, __float_as_uint(m));
}

__global__ __launch_bounds__(256) void k_quant(const float4* __restrict__ src,
                                               char4* __restrict__ dst, long n4,
                                               const unsigned* __restrict__ slot) {
  float inv = 127.f / fmaxf(__uint_as_float(*slot), 1e-8f);
  long i = (long)blockIdx.x * blockDim.x + threadIdx.x;
  long stride = (long)gridDim.x * blockDim.x;
  for (; i < n4; i += stride) {
    float4 v = src[i];
    char4 q;
    q.x = (signed char)__float2int_rn(v.x * inv);
    q.y = (signed char)__float2int_rn(v.y * inv);
    q.z = (signed char)__float2int_rn(v.z * inv);
    q.w = (signed char)__float2int_rn(v.w * inv);
    dst[i] = q;
  }
}

__device__ __forceinline__ v8i load_a_frag(const uint8_t* tile, int m0, int lane) {
  const uint8_t* p = tile + (m0 + (lane & 15)) * TSTRIDE + ((lane >> 4) << 4);
  unsigned long long d0 = *(const unsigned long long*)(p);
  unsigned long long d1 = *(const unsigned long long*)(p + 8);
  unsigned long long d2 = *(const unsigned long long*)(p + 32);
  unsigned long long d3 = *(const unsigned long long*)(p + 40);
  v8i a;
  a[0] = (int)(unsigned)d0; a[1] = (int)(d0 >> 32);
  a[2] = (int)(unsigned)d1; a[3] = (int)(d1 >> 32);
  a[4] = (int)(unsigned)d2; a[5] = (int)(d2 >> 32);
  a[6] = (int)(unsigned)d3; a[7] = (int)(d3 >> 32);
  return a;
}

__device__ __forceinline__ v8i load_b_frag(const uint8_t* tile, int n0, int lane) {
  const uint8_t* p = tile + (n0 + (lane & 15)) * TSTRIDE + ((lane >> 4) << 4);
  uint4 d0 = *(const uint4*)(p);
  uint4 d1 = *(const uint4*)(p + 32);
  v8i b;
  b[0] = (int)d0.x; b[1] = (int)d0.y; b[2] = (int)d0.z; b[3] = (int)d0.w;
  b[4] = (int)d1.x; b[5] = (int)d1.y; b[6] = (int)d1.z; b[7] = (int)d1.w;
  return b;
}

__global__ __launch_bounds__(256) void k_gemm1(
    const int8_t* __restrict__ qx, const int8_t* __restrict__ qw1,
    const int8_t* __restrict__ qw3, float* __restrict__ h,
    const unsigned* __restrict__ slots, unsigned* __restrict__ hmax_slot) {
  __shared__ __align__(16) uint8_t ldsA[2][128 * TSTRIDE];
  __shared__ __align__(16) uint8_t ldsB1[2][64 * TSTRIDE];
  __shared__ __align__(16) uint8_t ldsB3[2][64 * TSTRIDE];

  const int tid = threadIdx.x;
  const int lane = tid & 31, wid = tid >> 5;
  const int wm = wid & 1;
  const int wn = wid >> 1;
  const int blockF = blockIdx.x * 64;
  const int blockM = blockIdx.y * 128;

  const int rA0 = tid >> 2, rA1 = (tid >> 2) + 64;
  const int cC = (tid & 3) << 4;
  const int rB = tid >> 2;

  v8i acc1[4], acc3[4];
#pragma unroll
  for (int i = 0; i < 4; ++i) { acc1[i] = vzero8(); acc3[i] = vzero8(); }

  const int8_t* gA  = qx  + (size_t)blockM * D_DIM + cC;
  const int8_t* gB1 = qw1 + (size_t)blockF * D_DIM + cC;
  const int8_t* gB3 = qw3 + (size_t)blockF * D_DIM + cC;

#if HAVE_ASYNC_LDS
  async_cp16(ldsA[0]  + rA0 * TSTRIDE + cC, gA  + (size_t)rA0 * D_DIM);
  async_cp16(ldsA[0]  + rA1 * TSTRIDE + cC, gA  + (size_t)rA1 * D_DIM);
  async_cp16(ldsB1[0] + rB  * TSTRIDE + cC, gB1 + (size_t)rB  * D_DIM);
  async_cp16(ldsB3[0] + rB  * TSTRIDE + cC, gB3 + (size_t)rB  * D_DIM);
  wait_async0();
  __syncthreads();
  int p = 0;
  for (int k0 = 0; k0 < D_DIM; k0 += 64, p ^= 1) {
    if (k0 + 64 < D_DIM) {
      const int kn = k0 + 64;
      async_cp16(ldsA[p ^ 1]  + rA0 * TSTRIDE + cC, gA  + (size_t)rA0 * D_DIM + kn);
      async_cp16(ldsA[p ^ 1]  + rA1 * TSTRIDE + cC, gA  + (size_t)rA1 * D_DIM + kn);
      async_cp16(ldsB1[p ^ 1] + rB  * TSTRIDE + cC, gB1 + (size_t)rB  * D_DIM + kn);
      async_cp16(ldsB3[p ^ 1] + rB  * TSTRIDE + cC, gB3 + (size_t)rB  * D_DIM + kn);
    }
    v8i b1 = load_b_frag(ldsB1[p], wn * 16, lane);
    v8i b3 = load_b_frag(ldsB3[p], wn * 16, lane);
#pragma unroll
    for (int im = 0; im < 4; ++im) {
      v8i a = load_a_frag(ldsA[p], wm * 64 + im * 16, lane);
      acc1[im] = __builtin_amdgcn_wmma_i32_16x16x64_iu8(true, a, true, b1, acc1[im], false, false);
      acc3[im] = __builtin_amdgcn_wmma_i32_16x16x64_iu8(true, a, true, b3, acc3[im], false, false);
    }
    wait_async0();
    __syncthreads();
  }
#else
  uint4 vA0 = *(const uint4*)(gA + (size_t)rA0 * D_DIM);
  uint4 vA1 = *(const uint4*)(gA + (size_t)rA1 * D_DIM);
  uint4 vB1 = *(const uint4*)(gB1 + (size_t)rB * D_DIM);
  uint4 vB3 = *(const uint4*)(gB3 + (size_t)rB * D_DIM);
  *(uint4*)(ldsA[0]  + rA0 * TSTRIDE + cC) = vA0;
  *(uint4*)(ldsA[0]  + rA1 * TSTRIDE + cC) = vA1;
  *(uint4*)(ldsB1[0] + rB  * TSTRIDE + cC) = vB1;
  *(uint4*)(ldsB3[0] + rB  * TSTRIDE + cC) = vB3;
  __syncthreads();
  int p = 0;
  for (int k0 = 0; k0 < D_DIM; k0 += 64, p ^= 1) {
    const bool more = (k0 + 64 < D_DIM);
    if (more) {
      const int kn = k0 + 64;
      vA0 = *(const uint4*)(gA + (size_t)rA0 * D_DIM + kn);
      vA1 = *(const uint4*)(gA + (size_t)rA1 * D_DIM + kn);
      vB1 = *(const uint4*)(gB1 + (size_t)rB * D_DIM + kn);
      vB3 = *(const uint4*)(gB3 + (size_t)rB * D_DIM + kn);
    }
    v8i b1 = load_b_frag(ldsB1[p], wn * 16, lane);
    v8i b3 = load_b_frag(ldsB3[p], wn * 16, lane);
#pragma unroll
    for (int im = 0; im < 4; ++im) {
      v8i a = load_a_frag(ldsA[p], wm * 64 + im * 16, lane);
      acc1[im] = __builtin_amdgcn_wmma_i32_16x16x64_iu8(true, a, true, b1, acc1[im], false, false);
      acc3[im] = __builtin_amdgcn_wmma_i32_16x16x64_iu8(true, a, true, b3, acc3[im], false, false);
    }
    if (more) {
      *(uint4*)(ldsA[p ^ 1]  + rA0 * TSTRIDE + cC) = vA0;
      *(uint4*)(ldsA[p ^ 1]  + rA1 * TSTRIDE + cC) = vA1;
      *(uint4*)(ldsB1[p ^ 1] + rB  * TSTRIDE + cC) = vB1;
      *(uint4*)(ldsB3[p ^ 1] + rB  * TSTRIDE + cC) = vB3;
    }
    __syncthreads();
  }
#endif

  const float sx  = fmaxf(__uint_as_float(slots[0]), 1e-8f) / 127.f;
  const float sw1 = fmaxf(__uint_as_float(slots[1]), 1e-8f) / 127.f;
  const float sw3 = fmaxf(__uint_as_float(slots[2]), 1e-8f) / 127.f;
  const float s1 = sx * sw1, s3 = sx * sw3;

  float lmax = 0.f;
  const int col = blockF + wn * 16 + (lane & 15);
#pragma unroll
  for (int im = 0; im < 4; ++im) {
    const int rbase = blockM + wm * 64 + im * 16 + ((lane >> 4) << 3);
#pragma unroll
    for (int i = 0; i < 8; ++i) {
      float z3 = s3 * (float)acc3[im][i];
      float g  = z3 / (1.f + __expf(-z3));
      float hv = s1 * (float)acc1[im][i] * g;
      h[(size_t)(rbase + i) * F_DIM + col] = hv;
      lmax = fmaxf(lmax, fabsf(hv));
    }
  }
#pragma unroll
  for (int off = 16; off > 0; off >>= 1)
    lmax = fmaxf(lmax, __shfl_xor(lmax, off, 32));
  if (lane == 0) atomicMax(hmax_slot, __float_as_uint(lmax));
}

__global__ __launch_bounds__(256) void k_gemm2(
    const int8_t* __restrict__ qh, const int8_t* __restrict__ qw2,
    float* __restrict__ out, const unsigned* __restrict__ slots) {
  __shared__ __align__(16) uint8_t ldsA[2][128 * TSTRIDE];
  __shared__ __align__(16) uint8_t ldsB[2][128 * TSTRIDE];

  const int tid = threadIdx.x;
  const int lane = tid & 31, wid = tid >> 5;
  const int wm = wid & 1;
  const int wn = wid >> 1;
  const int blockD = blockIdx.x * 128;
  const int blockM = blockIdx.y * 128;

  const int r0 = tid >> 2, r1 = (tid >> 2) + 64;
  const int cC = (tid & 3) << 4;

  v8i acc[4][2];
#pragma unroll
  for (int i = 0; i < 4; ++i) { acc[i][0] = vzero8(); acc[i][1] = vzero8(); }

  const int8_t* gA = qh  + (size_t)blockM * F_DIM + cC;
  const int8_t* gB = qw2 + (size_t)blockD * F_DIM + cC;

#if HAVE_ASYNC_LDS
  async_cp16(ldsA[0] + r0 * TSTRIDE + cC, gA + (size_t)r0 * F_DIM);
  async_cp16(ldsA[0] + r1 * TSTRIDE + cC, gA + (size_t)r1 * F_DIM);
  async_cp16(ldsB[0] + r0 * TSTRIDE + cC, gB + (size_t)r0 * F_DIM);
  async_cp16(ldsB[0] + r1 * TSTRIDE + cC, gB + (size_t)r1 * F_DIM);
  wait_async0();
  __syncthreads();
  int p = 0;
  for (int k0 = 0; k0 < F_DIM; k0 += 64, p ^= 1) {
    if (k0 + 64 < F_DIM) {
      const int kn = k0 + 64;
      async_cp16(ldsA[p ^ 1] + r0 * TSTRIDE + cC, gA + (size_t)r0 * F_DIM + kn);
      async_cp16(ldsA[p ^ 1] + r1 * TSTRIDE + cC, gA + (size_t)r1 * F_DIM + kn);
      async_cp16(ldsB[p ^ 1] + r0 * TSTRIDE + cC, gB + (size_t)r0 * F_DIM + kn);
      async_cp16(ldsB[p ^ 1] + r1 * TSTRIDE + cC, gB + (size_t)r1 * F_DIM + kn);
    }
    v8i b0 = load_b_frag(ldsB[p], wn * 32, lane);
    v8i b1 = load_b_frag(ldsB[p], wn * 32 + 16, lane);
#pragma unroll
    for (int im = 0; im < 4; ++im) {
      v8i a = load_a_frag(ldsA[p], wm * 64 + im * 16, lane);
      acc[im][0] = __builtin_amdgcn_wmma_i32_16x16x64_iu8(true, a, true, b0, acc[im][0], false, false);
      acc[im][1] = __builtin_amdgcn_wmma_i32_16x16x64_iu8(true, a, true, b1, acc[im][1], false, false);
    }
    wait_async0();
    __syncthreads();
  }
#else
  uint4 vA0 = *(const uint4*)(gA + (size_t)r0 * F_DIM);
  uint4 vA1 = *(const uint4*)(gA + (size_t)r1 * F_DIM);
  uint4 vB0 = *(const uint4*)(gB + (size_t)r0 * F_DIM);
  uint4 vB1 = *(const uint4*)(gB + (size_t)r1 * F_DIM);
  *(uint4*)(ldsA[0] + r0 * TSTRIDE + cC) = vA0;
  *(uint4*)(ldsA[0] + r1 * TSTRIDE + cC) = vA1;
  *(uint4*)(ldsB[0] + r0 * TSTRIDE + cC) = vB0;
  *(uint4*)(ldsB[0] + r1 * TSTRIDE + cC) = vB1;
  __syncthreads();
  int p = 0;
  for (int k0 = 0; k0 < F_DIM; k0 += 64, p ^= 1) {
    const bool more = (k0 + 64 < F_DIM);
    if (more) {
      const int kn = k0 + 64;
      vA0 = *(const uint4*)(gA + (size_t)r0 * F_DIM + kn);
      vA1 = *(const uint4*)(gA + (size_t)r1 * F_DIM + kn);
      vB0 = *(const uint4*)(gB + (size_t)r0 * F_DIM + kn);
      vB1 = *(const uint4*)(gB + (size_t)r1 * F_DIM + kn);
    }
    v8i b0 = load_b_frag(ldsB[p], wn * 32, lane);
    v8i b1 = load_b_frag(ldsB[p], wn * 32 + 16, lane);
#pragma unroll
    for (int im = 0; im < 4; ++im) {
      v8i a = load_a_frag(ldsA[p], wm * 64 + im * 16, lane);
      acc[im][0] = __builtin_amdgcn_wmma_i32_16x16x64_iu8(true, a, true, b0, acc[im][0], false, false);
      acc[im][1] = __builtin_amdgcn_wmma_i32_16x16x64_iu8(true, a, true, b1, acc[im][1], false, false);
    }
    if (more) {
      *(uint4*)(ldsA[p ^ 1] + r0 * TSTRIDE + cC) = vA0;
      *(uint4*)(ldsA[p ^ 1] + r1 * TSTRIDE + cC) = vA1;
      *(uint4*)(ldsB[p ^ 1] + r0 * TSTRIDE + cC) = vB0;
      *(uint4*)(ldsB[p ^ 1] + r1 * TSTRIDE + cC) = vB1;
    }
    __syncthreads();
  }
#endif

  const float sh  = fmaxf(__uint_as_float(slots[4]), 1e-8f) / 127.f;
  const float sw2 = fmaxf(__uint_as_float(slots[3]), 1e-8f) / 127.f;
  const float s = sh * sw2;

#pragma unroll
  for (int im = 0; im < 4; ++im) {
    const int rbase = blockM + wm * 64 + im * 16 + ((lane >> 4) << 3);
#pragma unroll
    for (int jn = 0; jn < 2; ++jn) {
      const int col = blockD + wn * 32 + jn * 16 + (lane & 15);
#pragma unroll
      for (int i = 0; i < 8; ++i)
        out[(size_t)(rbase + i) * D_DIM + col] = s * (float)acc[im][jn][i];
    }
  }
}

extern "C" void kernel_launch(void* const* d_in, const int* in_sizes, int n_in,
                              void* d_out, int out_size, void* d_ws, size_t ws_size,
                              hipStream_t stream) {
  (void)in_sizes; (void)n_in; (void)out_size; (void)ws_size;
  const float* x  = (const float*)d_in[0];
  const float* w1 = (const float*)d_in[1];
  const float* w3 = (const float*)d_in[2];
  const float* w2 = (const float*)d_in[3];
  float* out = (float*)d_out;

  uint8_t* ws = (uint8_t*)d_ws;
  unsigned* slots = (unsigned*)ws;
  int8_t* qx  = (int8_t*)(ws + 256);
  int8_t* qw1 = qx  + (size_t)N_TOK * D_DIM;
  int8_t* qw3 = qw1 + (size_t)F_DIM * D_DIM;
  int8_t* qw2 = qw3 + (size_t)F_DIM * D_DIM;
  float*  h   = (float*)(qw2 + (size_t)F_DIM * D_DIM);
  int8_t* qh  = (int8_t*)(h + (size_t)N_TOK * F_DIM);

  const long nx4 = (long)N_TOK * D_DIM / 4;
  const long nw4 = (long)F_DIM * D_DIM / 4;
  const long nh4 = (long)N_TOK * F_DIM / 4;

  k_init_slots<<<1, 8, 0, stream>>>(slots);

  k_absmax<<<2048, 256, 0, stream>>>((const float4*)x,  nx4, slots + 0);
  k_absmax<<<2048, 256, 0, stream>>>((const float4*)w1, nw4, slots + 1);
  k_absmax<<<2048, 256, 0, stream>>>((const float4*)w3, nw4, slots + 2);
  k_absmax<<<2048, 256, 0, stream>>>((const float4*)w2, nw4, slots + 3);

  k_quant<<<4096, 256, 0, stream>>>((const float4*)x,  (char4*)qx,  nx4, slots + 0);
  k_quant<<<4096, 256, 0, stream>>>((const float4*)w1, (char4*)qw1, nw4, slots + 1);
  k_quant<<<4096, 256, 0, stream>>>((const float4*)w3, (char4*)qw3, nw4, slots + 2);
  k_quant<<<4096, 256, 0, stream>>>((const float4*)w2, (char4*)qw2, nw4, slots + 3);

  k_gemm1<<<dim3(F_DIM / 64, N_TOK / 128), 256, 0, stream>>>(qx, qw1, qw3, h, slots, slots + 4);

  k_quant<<<4096, 256, 0, stream>>>((const float4*)h, (char4*)qh, nh4, slots + 4);

  k_gemm2<<<dim3(D_DIM / 128, N_TOK / 128), 256, 0, stream>>>(qh, qw2, out, slots);
}
